// ResonanceMemoryAttentionV15_72172630442600
// MI455X (gfx1250) — matrix operand order repaired and checked by reading, not compiled, not run on hardware
//
#include <hip/hip_runtime.h>
#include <hip/hip_bf16.h>
#include <cmath>

typedef _Float16 h16;
typedef __attribute__((ext_vector_type(16))) _Float16 v16h;
typedef __attribute__((ext_vector_type(8)))  _Float16 h16x8;
typedef __attribute__((ext_vector_type(8)))  float    v8f;

#define DH      64
#define CHUNK   128
#define NCHUNK  8

// ---------------------------------------------------------------- utilities
// Convert a row-major f32 [K,N] weight matrix into f16 WMMA-fragment order:
// per 32(K)x16(N) tile, element e of lane l lives at tile*512 + l*16 + e and
// corresponds to B[(tk*32 + (l>>4)*8 + (e&7) + (e>>3)*16)*N + tn*16 + (l&15)].
__global__ void rma_pack_b(const float* __restrict__ src, h16* __restrict__ dst,
                           int K, int N) {
    int i = blockIdx.x * blockDim.x + threadIdx.x;
    if (i >= K * N) return;
    int e    = i & 15;
    int lane = (i >> 4) & 31;
    int tile = i >> 9;
    int ntiles = N >> 4;
    int tn = tile % ntiles;
    int tk = tile / ntiles;
    int kk  = tk * 32 + (lane >> 4) * 8 + (e & 7) + (e >> 3) * 16;
    int col = tn * 16 + (lane & 15);
    dst[i] = (h16)src[(size_t)kk * N + col];
}

// LayerNorm over D=768 (one row per 256-thread block, 3 elems/thread),
// emits fp16 x (for bottleneck GEMM) and fp16 x_norm (for QKV GEMM).
__global__ __launch_bounds__(256) void rma_ln_convert(
    const float* __restrict__ x, const float* __restrict__ w, const float* __restrict__ b,
    h16* __restrict__ xh, h16* __restrict__ xnh, int D)
{
    int row = blockIdx.x;
    int tid = threadIdx.x;
    const float* xr = x + (size_t)row * D;
    __shared__ float s1[8], s2[8];
    float vals[3];
    float lsum = 0.f, lsq = 0.f;
    #pragma unroll
    for (int i = 0; i < 3; ++i) {
        int c = tid + i * 256;
        float v = xr[c];
        vals[i] = v; lsum += v; lsq += v * v;
    }
    #pragma unroll
    for (int m = 16; m >= 1; m >>= 1) {
        lsum += __shfl_xor(lsum, m, 32);
        lsq  += __shfl_xor(lsq,  m, 32);
    }
    int wid = tid >> 5;
    if ((tid & 31) == 0) { s1[wid] = lsum; s2[wid] = lsq; }
    __syncthreads();
    float tot = 0.f, tot2 = 0.f;
    #pragma unroll
    for (int i = 0; i < 8; ++i) { tot += s1[i]; tot2 += s2[i]; }
    float mu  = tot / (float)D;
    float var = tot2 / (float)D - mu * mu;
    float rs  = rsqrtf(var + 1e-5f);
    #pragma unroll
    for (int i = 0; i < 3; ++i) {
        int c = tid + i * 256;
        xh [(size_t)row * D + c] = (h16)vals[i];
        xnh[(size_t)row * D + c] = (h16)((vals[i] - mu) * rs * w[c] + b[c]);
    }
}

// ---------------------------------------------------------------- WMMA GEMM
// C[M,N] = A[M,K](f16,row) @ B[K,N] where B is pre-packed in fragment order.
// Each wave computes a 32x64 strip: 2 A fragments x 4 B fragments = 8 WMMAs
// per K-step; all loads are contiguous 128-bit.  mode 0: f32 out + bias.
// mode 1: fused SiLU, f16 out.  Requires M%32==0, N%64==0, K%32==0.
__global__ __launch_bounds__(256) void rma_gemm_wmma(
    const h16* __restrict__ A, const h16* __restrict__ Bp,
    const float* __restrict__ bias, float* __restrict__ Cf, h16* __restrict__ Ch,
    int M, int N, int K, int mode)
{
    int wave = blockIdx.x * (blockDim.x >> 5) + (threadIdx.x >> 5);
    int nStrips = N >> 6;
    int totalWaves = (M >> 5) * nStrips;
    if (wave >= totalWaves) return;
    int mTile = wave / nStrips;          // 32 rows
    int nTile = wave % nStrips;          // 64 cols
    int lane    = threadIdx.x & 31;
    int halfSel = lane >> 4;             // 0: lanes 0-15, 1: lanes 16-31
    int l15     = lane & 15;
    int ntiles16 = N >> 4;

    v8f acc[2][4] = {};
    const h16* Arow0 = A + (size_t)(mTile * 32 + l15) * K;
    const h16* Arow1 = Arow0 + (size_t)16 * K;
    int aBase = halfSel * 8;             // K offset for this lane-half (A layout)
    const h16* Blane = Bp + (size_t)lane * 16;

    for (int k0 = 0; k0 < K; k0 += 32) {
        // A fragments: elems 0..7 at K = k0+aBase+., elems 8..15 at K = k0+16+aBase+.
        h16x8 lo0 = *(const h16x8*)(Arow0 + k0 + aBase);
        h16x8 hi0 = *(const h16x8*)(Arow0 + k0 + 16 + aBase);
        h16x8 lo1 = *(const h16x8*)(Arow1 + k0 + aBase);
        h16x8 hi1 = *(const h16x8*)(Arow1 + k0 + 16 + aBase);
        v16h a0 = __builtin_shufflevector(lo0, hi0, 0,1,2,3,4,5,6,7,8,9,10,11,12,13,14,15);
        v16h a1 = __builtin_shufflevector(lo1, hi1, 0,1,2,3,4,5,6,7,8,9,10,11,12,13,14,15);
        // B fragments: contiguous 512-element packed tiles
        const h16* Bt = Blane + ((size_t)(k0 >> 5) * ntiles16 + nTile * 4) * 512;
        #pragma unroll
        for (int sub = 0; sub < 4; ++sub) {
            v16h bfrag = *(const v16h*)(Bt + sub * 512);
            acc[0][sub] = __builtin_amdgcn_wmma_f32_16x16x32_f16(
                false, a0, false, bfrag, (short)0, acc[0][sub], false, false);
            acc[1][sub] = __builtin_amdgcn_wmma_f32_16x16x32_f16(
                false, a1, false, bfrag, (short)0, acc[1][sub], false, false);
        }
    }

    #pragma unroll
    for (int m = 0; m < 2; ++m) {
        #pragma unroll
        for (int sub = 0; sub < 4; ++sub) {
            int col = (nTile << 6) + (sub << 4) + l15;
            #pragma unroll
            for (int r = 0; r < 8; ++r) {
                int row = mTile * 32 + m * 16 + halfSel * 8 + r;   // C/D layout
                float v = acc[m][sub][r];
                if (mode == 0) {
                    Cf[(size_t)row * N + col] = v + bias[col];
                } else {
                    float s = v / (1.f + expf(-v));                 // SiLU
                    Ch[(size_t)row * N + col] = (h16)s;
                }
            }
        }
    }
}

// ---------------------------------------------------------------- gate path
// params = h1(f16) @ Wb2(f32)  -> interference gate.  One thread per (token, head).
__global__ void rma_gate(const h16* __restrict__ h1, const float* __restrict__ Wb2,
                         const float* __restrict__ temperature, float* __restrict__ gate,
                         int BL, int H)
{
    int gid = blockIdx.x * blockDim.x + threadIdx.x;
    if (gid >= BL * H) return;
    int row = gid / H, h = gid % H;
    float acc[5] = {0.f, 0.f, 0.f, 0.f, 0.f};
    const h16* hr = h1 + (size_t)row * 128;
    const float* wb = Wb2 + h * 5;
    for (int r = 0; r < 128; ++r) {
        float hv = (float)hr[r];
        const float* wr = wb + (size_t)r * (H * 5);
        #pragma unroll
        for (int j = 0; j < 5; ++j) acc[j] += hv * wr[j];
    }
    const float PI = 3.14159265358979323846f;
    float sa = 1.f / (1.f + expf(-acc[0]));
    float sp = tanhf(acc[1]) * PI;
    float ca = 1.f / (1.f + expf(-acc[2]));
    float cp = tanhf(acc[3]) * PI;
    float tv = fminf(fmaxf(temperature[0], 0.1f), 2.0f);
    float inter = tanhf(sa * ca * cosf(sp - cp)) * tv;
    gate[gid] = 1.f / (1.f + expf(-inter));
}

// -------------------------------------------------- chunked linear-attn scan
// Pass A: per-(b,h,chunk) sums of gated k v^T and k_f.  Thread e owns column e.
__global__ __launch_bounds__(64) void rma_chunk_sums(
    const float* __restrict__ qkv, const float* __restrict__ gate,
    float* __restrict__ cKV, float* __restrict__ cK, int L, int H, int D)
{
    int blk = blockIdx.x;                 // bh*NCHUNK + c
    int c  = blk % NCHUNK;
    int bh = blk / NCHUNK;
    int h  = bh % H;
    int b  = bh / H;
    int e  = threadIdx.x;
    __shared__ __align__(16) float kb[DH];
    float state[DH];
    #pragma unroll
    for (int d = 0; d < DH; ++d) state[d] = 0.f;
    float z = 0.f;
    for (int l = 0; l < CHUNK; ++l) {
        int row = b * L + c * CHUNK + l;
        size_t base = (size_t)row * 3 * D + h * DH + e;
        float k = qkv[base + D];
        float v = qkv[base + 2 * D];
        float kf = (k > 0.f) ? (k + 1.f) : expf(k);   // elu(k)+1
        float g  = 1.f + gate[(size_t)row * H + h];
        kb[e] = kf;
        __syncthreads();
        float vg = v * g;
        #pragma unroll
        for (int d = 0; d < DH; d += 4) {
            float4 kk = *(const float4*)&kb[d];
            state[d]   += kk.x * vg;
            state[d+1] += kk.y * vg;
            state[d+2] += kk.z * vg;
            state[d+3] += kk.w * vg;
        }
        z += kf;
        __syncthreads();
    }
    size_t cbase = (size_t)blk * DH * DH;
    #pragma unroll
    for (int d = 0; d < DH; ++d) cKV[cbase + (size_t)d * DH + e] = state[d];
    cK[(size_t)blk * DH + e] = z;
}

// Pass B: in-place exclusive prefix over the NCHUNK chunk sums per (b,h).
__global__ __launch_bounds__(64) void rma_chunk_scan(float* __restrict__ cKV,
                                                     float* __restrict__ cK)
{
    int bh = blockIdx.x;
    int e  = threadIdx.x;
    float run[DH];
    #pragma unroll
    for (int d = 0; d < DH; ++d) run[d] = 0.f;
    float zr = 0.f;
    for (int c = 0; c < NCHUNK; ++c) {
        size_t cbase = ((size_t)bh * NCHUNK + c) * DH * DH;
        #pragma unroll
        for (int d = 0; d < DH; ++d) {
            size_t idx = cbase + (size_t)d * DH + e;
            float t = cKV[idx];
            cKV[idx] = run[d];
            run[d] += t;
        }
        size_t kidx = ((size_t)bh * NCHUNK + c) * DH + e;
        float t = cK[kidx];
        cK[kidx] = zr;
        zr += t;
    }
}

// Pass C: within-chunk sequential pass, fused elu+1 / denominator / head LN.
__global__ __launch_bounds__(64) void rma_chunk_out(
    const float* __restrict__ qkv, const float* __restrict__ gate,
    const float* __restrict__ cKV, const float* __restrict__ cK,
    const float* __restrict__ mn_w, const float* __restrict__ mn_b,
    h16* __restrict__ attn, int L, int H, int D)
{
    int blk = blockIdx.x;
    int c  = blk % NCHUNK;
    int bh = blk / NCHUNK;
    int h  = bh % H;
    int b  = bh / H;
    int e  = threadIdx.x;
    int lane = e & 31, wid = e >> 5;
    __shared__ __align__(16) float kb[DH];
    __shared__ __align__(16) float qb[DH];
    __shared__ float sden[2], sy[2], sy2[2];

    float state[DH];
    size_t cbase = (size_t)blk * DH * DH;
    #pragma unroll
    for (int d = 0; d < DH; ++d) state[d] = cKV[cbase + (size_t)d * DH + e];
    float z  = cK[(size_t)blk * DH + e];
    float mw = mn_w[e], mb = mn_b[e];

    for (int l = 0; l < CHUNK; ++l) {
        int row = b * L + c * CHUNK + l;
        size_t base = (size_t)row * 3 * D + h * DH + e;
        float q = qkv[base];
        float k = qkv[base + D];
        float v = qkv[base + 2 * D];
        float qf = (q > 0.f) ? (q + 1.f) : expf(q);
        float kf = (k > 0.f) ? (k + 1.f) : expf(k);
        float g  = 1.f + gate[(size_t)row * H + h];
        kb[e] = kf; qb[e] = qf;
        __syncthreads();                                     // A
        float vg = v * g;
        float num = 0.f;
        #pragma unroll
        for (int d = 0; d < DH; d += 4) {
            float4 kk = *(const float4*)&kb[d];
            float4 qq = *(const float4*)&qb[d];
            state[d]   += kk.x * vg; num += qq.x * state[d];
            state[d+1] += kk.y * vg; num += qq.y * state[d+1];
            state[d+2] += kk.z * vg; num += qq.z * state[d+2];
            state[d+3] += kk.w * vg; num += qq.w * state[d+3];
        }
        z += kf;
        float pden = qf * z;
        #pragma unroll
        for (int m = 16; m >= 1; m >>= 1) pden += __shfl_xor(pden, m, 32);
        if (lane == 0) sden[wid] = pden;
        __syncthreads();                                     // B
        float den = sden[0] + sden[1] + 1e-6f;
        float y = num / den;
        float py = y, py2 = y * y;
        #pragma unroll
        for (int m = 16; m >= 1; m >>= 1) {
            py  += __shfl_xor(py,  m, 32);
            py2 += __shfl_xor(py2, m, 32);
        }
        if (lane == 0) { sy[wid] = py; sy2[wid] = py2; }
        __syncthreads();                                     // C
        float mu  = (sy[0] + sy[1]) * (1.f / DH);
        float var = (sy2[0] + sy2[1]) * (1.f / DH) - mu * mu;
        float rs  = rsqrtf(var + 1e-5f);
        attn[(size_t)row * D + h * DH + e] = (h16)((y - mu) * rs * mw + mb);
    }
}

// ---------------------------------------------------------------- launcher
extern "C" void kernel_launch(void* const* d_in, const int* in_sizes, int n_in,
                              void* d_out, int out_size, void* d_ws, size_t ws_size,
                              hipStream_t stream) {
    (void)in_sizes; (void)n_in; (void)out_size; (void)ws_size;
    constexpr int B = 2, L = 1024, D = 768, H = 12, R = 128;
    constexpr int BL = B * L, D3 = 3 * D;

    const float* x     = (const float*)d_in[0];
    const float* Wqkv  = (const float*)d_in[1];
    const float* bqkv  = (const float*)d_in[2];
    const float* Wb1   = (const float*)d_in[3];
    const float* Wb2   = (const float*)d_in[4];
    const float* temp  = (const float*)d_in[5];
    const float* Wproj = (const float*)d_in[6];
    const float* bproj = (const float*)d_in[7];
    const float* ln_w  = (const float*)d_in[8];
    const float* ln_b  = (const float*)d_in[9];
    const float* mn_w  = (const float*)d_in[10];
    const float* mn_b  = (const float*)d_in[11];

    float* out_proj = (float*)d_out;
    float* out_gate = out_proj + (size_t)BL * D;

    char* ws = (char*)d_ws;
    size_t off = 0;
    auto alloc = [&](size_t bytes) -> void* {
        void* p = ws + off;
        off = (off + bytes + 255) & ~(size_t)255;
        return p;
    };
    h16*   xh      = (h16*)  alloc((size_t)BL * D * sizeof(h16));
    h16*   xnh     = (h16*)  alloc((size_t)BL * D * sizeof(h16));
    h16*   Wqkv_p  = (h16*)  alloc((size_t)D * D3 * sizeof(h16));
    h16*   Wb1_p   = (h16*)  alloc((size_t)D * R * sizeof(h16));
    h16*   Wproj_p = (h16*)  alloc((size_t)D * D * sizeof(h16));
    float* qkv     = (float*)alloc((size_t)BL * D3 * sizeof(float));
    h16*   h1      = (h16*)  alloc((size_t)BL * R * sizeof(h16));
    float* cKV     = (float*)alloc((size_t)B * H * NCHUNK * DH * DH * sizeof(float));
    float* cK      = (float*)alloc((size_t)B * H * NCHUNK * DH * sizeof(float));
    h16*   attn    = (h16*)  alloc((size_t)BL * D * sizeof(h16));

    // 1. weight conversions (fp32 -> fp16, WMMA-fragment packed)
    { int n = D * D3; rma_pack_b<<<(n + 255) / 256, 256, 0, stream>>>(Wqkv,  Wqkv_p,  D, D3); }
    { int n = D * R;  rma_pack_b<<<(n + 255) / 256, 256, 0, stream>>>(Wb1,   Wb1_p,   D, R);  }
    { int n = D * D;  rma_pack_b<<<(n + 255) / 256, 256, 0, stream>>>(Wproj, Wproj_p, D, D);  }

    // 2. LayerNorm(x) + fp16 converts
    rma_ln_convert<<<BL, 256, 0, stream>>>(x, ln_w, ln_b, xh, xnh, D);

    // 3. qkv = x_norm @ Wqkv + bqkv   (WMMA)
    { int waves = (BL / 32) * (D3 / 64);
      rma_gemm_wmma<<<(waves + 7) / 8, 256, 0, stream>>>(xnh, Wqkv_p, bqkv, qkv, nullptr, BL, D3, D, 0); }

    // 4. h1 = silu(x @ Wb1)           (WMMA, fused silu -> fp16)
    { int waves = (BL / 32) * (R / 64);
      rma_gemm_wmma<<<(waves + 7) / 8, 256, 0, stream>>>(xh, Wb1_p, nullptr, nullptr, h1, BL, R, D, 1); }

    // 5. interference gate (also the second output)
    rma_gate<<<(BL * H + 255) / 256, 256, 0, stream>>>(h1, Wb2, temp, out_gate, BL, H);

    // 6-8. chunked causal linear-attention scan
    rma_chunk_sums<<<B * H * NCHUNK, 64, 0, stream>>>(qkv, out_gate, cKV, cK, L, H, D);
    rma_chunk_scan<<<B * H, 64, 0, stream>>>(cKV, cK);
    rma_chunk_out <<<B * H * NCHUNK, 64, 0, stream>>>(qkv, out_gate, cKV, cK, mn_w, mn_b, attn, L, H, D);

    // 9. out = attn @ Wproj + bproj   (WMMA)
    { int waves = (BL / 32) * (D / 64);
      rma_gemm_wmma<<<(waves + 7) / 8, 256, 0, stream>>>(attn, Wproj_p, bproj, out_proj, nullptr, BL, D, D, 0); }
}
